// Contrastive_75204877353644
// MI455X (gfx1250) — matrix operand order repaired and checked by reading, not compiled, not run on hardware
//
#include <hip/hip_runtime.h>
#include <hip/hip_bf16.h>

typedef __attribute__((ext_vector_type(16))) _Float16 v16h;
typedef __attribute__((ext_vector_type(8)))  _Float16 v8h;
typedef __attribute__((ext_vector_type(8)))  float    v8f;
typedef __attribute__((ext_vector_type(4)))  float    v4f;

#define NROWS 8192
#define KDIM  2048
#define NPROJ 256
#define HALFB 4096
#define CSPLIT 8
// 10 / ln(2):  exp(x/0.1) = exp2(x * 10*log2(e))
#define SC10  14.426950408889634f
#define LN2F  0.6931471805599453f

__device__ __forceinline__ float fexp2(float x) {
#if __has_builtin(__builtin_amdgcn_exp2f)
    return __builtin_amdgcn_exp2f(x);
#else
    return exp2f(x);
#endif
}
__device__ __forceinline__ float flog2(float x) {
#if __has_builtin(__builtin_amdgcn_logf)
    return __builtin_amdgcn_logf(x);
#else
    return log2f(x);
#endif
}

__device__ __forceinline__ v8f wmma_f16(v16h a, v16h b, v8f c) {
    return __builtin_amdgcn_wmma_f32_16x16x32_f16(
        /*neg_a=*/false, a, /*neg_b=*/false, b,
        /*c_mod=*/(short)0, c, /*reuse_a=*/false, /*reuse_b=*/false);
}

// ---------------------------------------------------------------------------
// K1: W [2048,256] f32 -> WT [256,2048] f16   (B-fragments become contiguous)
// ---------------------------------------------------------------------------
__global__ void k_convert_wt(const float* __restrict__ W, _Float16* __restrict__ WT) {
    int idx = blockIdx.x * blockDim.x + threadIdx.x;      // 524288 threads
    int n = idx >> 11;          // 0..255
    int k = idx & (KDIM - 1);   // 0..2047
    WT[(size_t)n * KDIM + k] = (_Float16)W[(size_t)k * NPROJ + n];
}

// ---------------------------------------------------------------------------
// K2: out_head[8192,256] = A[8192,2048] @ W + b   (f16 WMMA, f32 accum)
// one wave -> one 16x16 tile, 8 waves / block
// ---------------------------------------------------------------------------
__global__ void k_gemm1(const float* __restrict__ A, const _Float16* __restrict__ WT,
                        const float* __restrict__ bias, float* __restrict__ outHead) {
    int wid  = blockIdx.x * 8 + (threadIdx.x >> 5);   // 0..8191
    int lane = threadIdx.x & 31;
    int lh = lane & 15, hb = lane >> 4;

    int colt = wid & 15;       // 16 column tiles
    int rowt = wid >> 4;       // 512 row tiles
    int rowbase = rowt * 16;
    int nbase   = colt * 16;

    const float*    arow = A  + (size_t)(rowbase + lh) * KDIM;
    const _Float16* brow = WT + (size_t)(nbase + lh) * KDIM;

    v8f c = {};
    for (int k0 = 0; k0 < KDIM; k0 += 32) {
        int ka = k0 + hb * 8;
        v4f f0 = *(const v4f*)(arow + ka);
        v4f f1 = *(const v4f*)(arow + ka + 4);
        v4f f2 = *(const v4f*)(arow + ka + 16);
        v4f f3 = *(const v4f*)(arow + ka + 20);
        v16h a;
#pragma unroll
        for (int t = 0; t < 4; ++t) {
            a[t]      = (_Float16)f0[t];
            a[4 + t]  = (_Float16)f1[t];
            a[8 + t]  = (_Float16)f2[t];
            a[12 + t] = (_Float16)f3[t];
        }
        v16h b = __builtin_shufflevector(*(const v8h*)(brow + k0 + hb * 8), *(const v8h*)(brow + k0 + hb * 8 + 16), 0, 1, 2, 3, 4, 5, 6, 7, 8, 9, 10, 11, 12, 13, 14, 15);
        c = wmma_f16(a, b, c);
    }
    float bv = bias[nbase + lh];
#pragma unroll
    for (int r = 0; r < 8; ++r) {
        int row = rowbase + r + hb * 8;
        outHead[(size_t)row * NPROJ + nbase + lh] = c[r] + bv;
    }
}

// ---------------------------------------------------------------------------
// K3: row-wise L2 normalize (f32) -> f16 matrix.  One wave per row.
// ---------------------------------------------------------------------------
__global__ void k_rownorm(const float* __restrict__ outHead, _Float16* __restrict__ outH) {
    int row  = blockIdx.x * 8 + (threadIdx.x >> 5);
    int lane = threadIdx.x & 31;
    const float* rp = outHead + (size_t)row * NPROJ;

    v8f x = *(const v8f*)(rp + lane * 8);
    float ss = 0.f;
#pragma unroll
    for (int t = 0; t < 8; ++t) ss += x[t] * x[t];
    ss += __shfl_xor(ss, 1, 32);
    ss += __shfl_xor(ss, 2, 32);
    ss += __shfl_xor(ss, 4, 32);
    ss += __shfl_xor(ss, 8, 32);
    ss += __shfl_xor(ss, 16, 32);
    float inv = 1.0f / fmaxf(sqrtf(ss), 1e-12f);

    v8h y;
#pragma unroll
    for (int t = 0; t < 8; ++t) y[t] = (_Float16)(x[t] * inv);
    *(v8h*)(outH + (size_t)row * NPROJ + lane * 8) = y;
}

// ---------------------------------------------------------------------------
// K4: per pair i<4096: selfexp for rows i and i+4096 (diagonal removal term)
//     and positive-pair natural log-numerator  poslog[i] = 10 * (a1 . a2)
// ---------------------------------------------------------------------------
__global__ void k_posself(const _Float16* __restrict__ outH,
                          float* __restrict__ selfexp, float* __restrict__ poslog) {
    int i = blockIdx.x * blockDim.x + threadIdx.x;   // 4096 threads
    const _Float16* r1 = outH + (size_t)i * NPROJ;
    const _Float16* r2 = outH + (size_t)(i + HALFB) * NPROJ;
    float s1 = 0.f, s2 = 0.f, p = 0.f;
    for (int k = 0; k < NPROJ; ++k) {
        float a = (float)r1[k];
        float b = (float)r2[k];
        s1 += a * a;
        s2 += b * b;
        p  += a * b;
    }
    selfexp[i]         = fexp2(s1 * SC10);
    selfexp[i + HALFB] = fexp2(s2 * SC10);
    poslog[i]          = p * 10.0f;
}

// ---------------------------------------------------------------------------
// K5: fused sim @ exp @ row-sum.  Wave owns 16 rows; streams a 1024-col split.
//     negpart[cs*8192 + row] = sum_{cols in split} exp(sim/T)
// ---------------------------------------------------------------------------
__global__ void k_simexp(const _Float16* __restrict__ outH, float* __restrict__ negpart) {
    int wid  = blockIdx.x * 8 + (threadIdx.x >> 5);  // 0..4095
    int lane = threadIdx.x & 31;
    int lh = lane & 15, hb = lane >> 4;

    int rowt = wid & 511;      // 512 row tiles
    int cs   = wid >> 9;       // 8 column splits
    int rowbase = rowt * 16;

    const _Float16* arow = outH + (size_t)(rowbase + lh) * NPROJ;

    // preload 8 A-fragments covering K = 0..255
    v16h a[8];
#pragma unroll
    for (int kk = 0; kk < 8; ++kk) {
        int base = kk * 32 + hb * 8;
        v8h lo = *(const v8h*)(arow + base);
        v8h hi = *(const v8h*)(arow + base + 16);
        a[kk] = __builtin_shufflevector(lo, hi, 0, 1, 2, 3, 4, 5, 6, 7,
                                        8, 9, 10, 11, 12, 13, 14, 15);
    }

    v8f acc = {};
    int nt0 = cs * 64;                         // 64 N-tiles of 16 columns = 1024 cols
    for (int nt = nt0; nt < nt0 + 64; ++nt) {
        const _Float16* brow = outH + (size_t)(nt * 16 + lh) * NPROJ;
        v8f c = {};
#pragma unroll
        for (int kk = 0; kk < 8; ++kk) {
            v16h b = __builtin_shufflevector(*(const v8h*)(brow + kk * 32 + hb * 8), *(const v8h*)(brow + kk * 32 + hb * 8 + 16), 0, 1, 2, 3, 4, 5, 6, 7, 8, 9, 10, 11, 12, 13, 14, 15);
            c = wmma_f16(a[kk], b, c);
        }
#pragma unroll
        for (int r = 0; r < 8; ++r) acc[r] += fexp2(c[r] * SC10);
    }

    // reduce across the 16 lanes of each half-wave (xor masks < 16 stay in-half)
#pragma unroll
    for (int r = 0; r < 8; ++r) {
        float v = acc[r];
        v += __shfl_xor(v, 1, 32);
        v += __shfl_xor(v, 2, 32);
        v += __shfl_xor(v, 4, 32);
        v += __shfl_xor(v, 8, 32);
        if (lh == 0) {
            int row = rowbase + r + hb * 8;
            negpart[(size_t)cs * NROWS + row] = v;   // unique (cs,row) writer
        }
    }
}

// ---------------------------------------------------------------------------
// K6: loss = -mean( poslog[i%4096] - ln( sum_cs negpart - selfexp ) )
// ---------------------------------------------------------------------------
__global__ void k_finalize(const float* __restrict__ negpart,
                           const float* __restrict__ selfexp,
                           const float* __restrict__ poslog,
                           float* __restrict__ out) {
    __shared__ float sm[256];
    int tid = threadIdx.x;
    float s = 0.f;
    for (int i = tid; i < NROWS; i += 256) {
        float neg = -selfexp[i];
#pragma unroll
        for (int c = 0; c < CSPLIT; ++c) neg += negpart[(size_t)c * NROWS + i];
        s += poslog[i & (HALFB - 1)] - flog2(neg) * LN2F;
    }
    sm[tid] = s;
    __syncthreads();
    for (int off = 128; off > 0; off >>= 1) {
        if (tid < off) sm[tid] += sm[tid + off];
        __syncthreads();
    }
    if (tid == 0) out[0] = -sm[0] / (float)NROWS;
}

// ---------------------------------------------------------------------------
extern "C" void kernel_launch(void* const* d_in, const int* in_sizes, int n_in,
                              void* d_out, int out_size, void* d_ws, size_t ws_size,
                              hipStream_t stream) {
    (void)in_sizes; (void)n_in; (void)out_size; (void)ws_size;
    const float* embedded = (const float*)d_in[0];   // [8192, 2048]
    const float* W        = (const float*)d_in[1];   // [2048, 256]
    const float* b        = (const float*)d_in[2];   // [256]
    float* out            = (float*)d_out;           // scalar loss

    char* ws = (char*)d_ws;
    _Float16* WT      = (_Float16*)(ws);                                  // 1 MB
    float*    outHead = (float*)(ws + (size_t)1048576);                   // 8 MB
    _Float16* outH    = (_Float16*)(ws + (size_t)1048576 + 8388608);      // 4 MB
    float*    negpart = (float*)(ws + (size_t)13631488);                  // 256 KB
    float*    selfexp = (float*)(ws + (size_t)13631488 + 262144);         // 32 KB
    float*    poslog  = (float*)(ws + (size_t)13631488 + 262144 + 32768); // 16 KB

    k_convert_wt<<<(KDIM * NPROJ) / 256, 256, 0, stream>>>(W, WT);
    k_gemm1<<<1024, 256, 0, stream>>>(embedded, WT, b, outHead);
    k_rownorm<<<NROWS / 8, 256, 0, stream>>>(outHead, outH);
    k_posself<<<HALFB / 256, 256, 0, stream>>>(outH, selfexp, poslog);
    k_simexp<<<512, 256, 0, stream>>>(outH, negpart);
    k_finalize<<<1, 256, 0, stream>>>(negpart, selfexp, poslog, out);
}
